// CGM_11381663335003
// MI455X (gfx1250) — compile-verified, matrix operand order checked by reading, not run on hardware
//
#include <hip/hip_runtime.h>
#include <math.h>

#define NHEAD 8
#define DHEAD 32
#define FDIM  256
#define KDIM  256

typedef __attribute__((ext_vector_type(16))) __bf16 v16bf;
typedef __attribute__((ext_vector_type(8)))  float  v8f;

struct alignas(16) B128 { unsigned int u[4]; };
union AB { B128 q[2]; v16bf v; };

__device__ __forceinline__ unsigned short f32_to_bf16_rne(float f) {
  unsigned u = __float_as_uint(f);
  unsigned r = u + 0x7FFFu + ((u >> 16) & 1u);
  return (unsigned short)(r >> 16);
}

// monotonic float<->uint encoding for atomicMax-based segment max
__device__ __forceinline__ unsigned enc_f(float x) {
  unsigned b = __float_as_uint(x);
  return (b & 0x80000000u) ? ~b : (b | 0x80000000u);
}
__device__ __forceinline__ float dec_f(unsigned u) {
  return (u & 0x80000000u) ? __uint_as_float(u & 0x7FFFFFFFu) : __uint_as_float(~u);
}

// ---------------- conversion kernels ----------------
__global__ void cvt_f32_bf16(const float* __restrict__ x, unsigned short* __restrict__ y, int n) {
  int i = blockIdx.x * blockDim.x + threadIdx.x;
  if (i < n) y[i] = f32_to_bf16_rne(x[i]);
}

// W[K,N] f32 (row major) -> Wt[N,K] bf16 (transposed)
__global__ void transpose_cvt(const float* __restrict__ W, unsigned short* __restrict__ Wt,
                              int K, int N) {
  int i = blockIdx.x * blockDim.x + threadIdx.x;
  if (i >= K * N) return;
  int k = i / N, n = i % N;
  Wt[(size_t)n * K + k] = f32_to_bf16_rne(W[(size_t)k * N + n]);
}

// ---------------- WMMA bf16 GEMM: C[M,N] = A[M,K] * Bt[N,K]^T ----------------
// one wave per 16x16 output tile; K multiple of 32; M,N multiples of 16
__global__ void wmma_gemm_bf16(const unsigned short* __restrict__ A,
                               const unsigned short* __restrict__ Bt,
                               float* __restrict__ C, int M, int N, int K) {
  int lane   = threadIdx.x & 31;
  int waveId = threadIdx.x >> 5;
  int wavesPerBlock = blockDim.x >> 5;
  int tile = blockIdx.x * wavesPerBlock + waveId;
  int tilesN = N >> 4, tilesM = M >> 4;
  if (tile >= tilesM * tilesN) return;          // wave-uniform exit, EXEC stays full
  int tm = tile / tilesN, tn = tile % tilesN;

  int lo  = lane & 15;          // row of A / col of B within tile
  int hi  = lane >> 4;          // half-wave select
  int kb  = hi << 3;            // 0 or 8
  size_t aRow = (size_t)((tm << 4) + lo) * K;
  size_t bRow = (size_t)((tn << 4) + lo) * K;

  v8f acc = {0.f,0.f,0.f,0.f,0.f,0.f,0.f,0.f};
  for (int k0 = 0; k0 < K; k0 += 32) {
    AB ua, ub;
    const B128* ap = reinterpret_cast<const B128*>(A + aRow + k0 + kb);
    const B128* bp = reinterpret_cast<const B128*>(Bt + bRow + k0 + kb);
    ua.q[0] = ap[0]; ua.q[1] = ap[2];   // K = k0+kb..+7  and  k0+kb+16..+23
    ub.q[0] = bp[0]; ub.q[1] = bp[2];
    acc = __builtin_amdgcn_wmma_f32_16x16x32_bf16(false, ua.v, false, ub.v,
                                                  (short)0, acc, false, false);
  }
  int r0 = hi << 3;
  int col = (tn << 4) + lo;
  #pragma unroll
  for (int r = 0; r < 8; ++r)
    C[(size_t)((tm << 4) + r0 + r) * N + col] = acc[r];
}

// ---------------- GAT edge pipeline ----------------
// e[n,h] = sum_d f[n,h*D+d] * a[h*D+d]
__global__ void attn_score(const float* __restrict__ f, const float* __restrict__ a,
                           float* __restrict__ e, int Nn) {
  int i = blockIdx.x * blockDim.x + threadIdx.x;
  if (i >= Nn * NHEAD) return;
  int n = i / NHEAD, h = i % NHEAD;
  const float* fp = f + (size_t)n * FDIM + h * DHEAD;
  const float* ap = a + h * DHEAD;
  float s = 0.f;
  #pragma unroll
  for (int d = 0; d < DHEAD; ++d) s += fp[d] * ap[d];
  e[i] = s;
}

__global__ void edge_max(const int* __restrict__ src, const int* __restrict__ dst,
                         const float* __restrict__ el, const float* __restrict__ er,
                         unsigned* __restrict__ emax, int E) {
  int i = blockIdx.x * blockDim.x + threadIdx.x;
  if (i >= E * NHEAD) return;
  int e = i / NHEAD, h = i % NHEAD;
  float x = el[(size_t)src[e] * NHEAD + h] + er[(size_t)dst[e] * NHEAD + h];
  x = (x > 0.f) ? x : 0.2f * x;                 // leaky_relu 0.2
  atomicMax(&emax[(size_t)dst[e] * NHEAD + h], enc_f(x));
}

__global__ void edge_exp(const int* __restrict__ src, const int* __restrict__ dst,
                         const float* __restrict__ el, const float* __restrict__ er,
                         const unsigned* __restrict__ emax, float* __restrict__ den,
                         float* __restrict__ ee, int E) {
  int i = blockIdx.x * blockDim.x + threadIdx.x;
  if (i >= E * NHEAD) return;
  int e = i / NHEAD, h = i % NHEAD;
  int d = dst[e];
  float x = el[(size_t)src[e] * NHEAD + h] + er[(size_t)d * NHEAD + h];
  x = (x > 0.f) ? x : 0.2f * x;
  float v = __expf(x - dec_f(emax[(size_t)d * NHEAD + h]));
  ee[i] = v;
  atomicAdd(&den[(size_t)d * NHEAD + h], v);
}

__global__ void edge_aggr(const int* __restrict__ src, const int* __restrict__ dst,
                          const float* __restrict__ ee, const float* __restrict__ den,
                          const float* __restrict__ fs, float* __restrict__ out, int E) {
  int i = blockIdx.x * blockDim.x + threadIdx.x;
  if (i >= E * NHEAD) return;
  int e = i / NHEAD, h = i % NHEAD;
  int s = src[e], d = dst[e];
  float coeff = ee[i] / den[(size_t)d * NHEAD + h];
  const float* fp = fs + (size_t)s * FDIM + h * DHEAD;
  float* op = out + (size_t)d * FDIM + h * DHEAD;
  #pragma unroll
  for (int k = 0; k < DHEAD; ++k) atomicAdd(&op[k], coeff * fp[k]);
}

__global__ void bias_elu(float* __restrict__ out, const float* __restrict__ bias, int Nn) {
  int i = blockIdx.x * blockDim.x + threadIdx.x;
  if (i >= Nn * FDIM) return;
  float x = out[i] + bias[i % FDIM];
  out[i] = (x > 0.f) ? x : (__expf(x) - 1.0f);   // elu, alpha=1
}

// ---------------- semantic attention (M=2 metapaths) ----------------
__global__ void sem_combine(const float* __restrict__ sA, const float* __restrict__ sB,
                            const float* __restrict__ b1, const float* __restrict__ w2,
                            const float* __restrict__ oA, const float* __restrict__ oB,
                            float* __restrict__ out, int Nn) {
  int n = blockIdx.x * blockDim.x + threadIdx.x;
  if (n >= Nn) return;
  float wA = 0.f, wB = 0.f;
  for (int j = 0; j < 128; ++j) {
    float bj = b1[j], wj = w2[j];
    wA += tanhf(sA[(size_t)n * 128 + j] + bj) * wj;
    wB += tanhf(sB[(size_t)n * 128 + j] + bj) * wj;
  }
  float m = fmaxf(wA, wB);
  float eA = __expf(wA - m), eB = __expf(wB - m);
  float inv = 1.0f / (eA + eB);
  float bAw = eA * inv, bBw = eB * inv;
  for (int f = 0; f < FDIM; ++f)
    out[(size_t)n * FDIM + f] = bAw * oA[(size_t)n * FDIM + f] + bBw * oB[(size_t)n * FDIM + f];
}

// ---------------- DNN head tail: g1[N,32] -> leaky -> 16 -> leaky -> 1 ----------------
__global__ void head_kernel(const float* __restrict__ g1, const float* __restrict__ b1,
                            const float* __restrict__ W2, const float* __restrict__ b2,
                            const float* __restrict__ W3, const float* __restrict__ b3,
                            float* __restrict__ out, int Nn) {
  int n = blockIdx.x * blockDim.x + threadIdx.x;
  if (n >= Nn) return;
  float h1[32];
  #pragma unroll
  for (int j = 0; j < 32; ++j) {
    float v = g1[(size_t)n * 32 + j] + b1[j];
    h1[j] = (v > 0.f) ? v : 0.01f * v;
  }
  float o = b3[0];
  #pragma unroll
  for (int j = 0; j < 16; ++j) {
    float v = b2[j];
    #pragma unroll
    for (int i = 0; i < 32; ++i) v += h1[i] * W2[i * 16 + j];
    v = (v > 0.f) ? v : 0.01f * v;
    o += v * W3[j];
  }
  out[n] = o;
}

static inline int cdiv_ll(long long a, long long b) { return (int)((a + b - 1) / b); }

extern "C" void kernel_launch(void* const* d_in, const int* in_sizes, int n_in,
                              void* d_out, int out_size, void* d_ws, size_t ws_size,
                              hipStream_t stream) {
  (void)n_in; (void)out_size; (void)ws_size;
  const float* feat_drug = (const float*)d_in[0];
  const float* feat_cell = (const float*)d_in[1];
  const int* src_dd = (const int*)d_in[2];
  const int* dst_dd = (const int*)d_in[3];
  const int* src_dc = (const int*)d_in[4];
  const int* dst_dc = (const int*)d_in[5];
  const int* src_cc = (const int*)d_in[6];
  const int* dst_cc = (const int*)d_in[7];
  const float* Wsrc    = (const float*)d_in[8];   // [2,3,256,256]
  const float* Wdst_dc = (const float*)d_in[9];   // [2,256,256]
  const float* attn_l  = (const float*)d_in[10];  // [2,3,H,D]
  const float* attn_r  = (const float*)d_in[11];
  const float* gat_b   = (const float*)d_in[12];  // [2,3,256]
  const float* sem_W1  = (const float*)d_in[13];  // [2,2,256,128]
  const float* sem_b1  = (const float*)d_in[14];  // [2,2,128]
  const float* sem_W2  = (const float*)d_in[15];  // [2,2,128]
  const float* dnn_W1  = (const float*)d_in[16];
  const float* dnn_b1  = (const float*)d_in[17];
  const float* dnn_W2  = (const float*)d_in[18];
  const float* dnn_b2  = (const float*)d_in[19];
  const float* dnn_W3  = (const float*)d_in[20];
  const float* dnn_b3  = (const float*)d_in[21];

  const int Nd  = in_sizes[0] / FDIM;
  const int Nc  = in_sizes[1] / FDIM;
  const int Edd = in_sizes[2];
  const int Edc = in_sizes[4];
  const int Ecc = in_sizes[6];
  int Emax = Edd > Edc ? Edd : Edc; if (Ecc > Emax) Emax = Ecc;

  // ---- workspace carve ----
  char* p = (char*)d_ws;
  auto carve = [&](size_t bytes) -> void* {
    void* r = (void*)p; p += (bytes + 255) & ~(size_t)255; return r;
  };
  unsigned short* bfXd = (unsigned short*)carve((size_t)Nd * FDIM * 2);
  unsigned short* bfXc = (unsigned short*)carve((size_t)Nc * FDIM * 2);
  unsigned short* Wt   = (unsigned short*)carve((size_t)KDIM * FDIM * 2);
  float* fsbuf = (float*)carve((size_t)Nd * FDIM * 4);
  float* fdbuf = (float*)carve((size_t)Nc * FDIM * 4);
  float* el    = (float*)carve((size_t)Nd * NHEAD * 4);
  float* er    = (float*)carve((size_t)Nd * NHEAD * 4);
  unsigned* emax = (unsigned*)carve((size_t)Nd * NHEAD * 4);
  float* den   = (float*)carve((size_t)Nd * NHEAD * 4);
  float* ee    = (float*)carve((size_t)Emax * NHEAD * 4);
  float* o_dd  = (float*)carve((size_t)Nd * FDIM * 4);
  float* o_dc  = (float*)carve((size_t)Nc * FDIM * 4);
  float* o_cc  = (float*)carve((size_t)Nc * FDIM * 4);
  float* hc1   = (float*)carve((size_t)Nc * FDIM * 4);
  float* hc2   = (float*)carve((size_t)Nc * FDIM * 4);
  float* sA    = (float*)carve((size_t)Nc * 128 * 4);
  float* sB    = (float*)carve((size_t)Nc * 128 * 4);
  unsigned short* bfA = (unsigned short*)carve((size_t)Nc * FDIM * 2);
  unsigned short* bfB = (unsigned short*)carve((size_t)Nc * FDIM * 2);
  float* g1    = (float*)carve((size_t)Nc * 32 * 4);

  // ---- launch helpers ----
  auto cvt = [&](const float* x, unsigned short* y, long long n) {
    cvt_f32_bf16<<<cdiv_ll(n, 256), 256, 0, stream>>>(x, y, (int)n);
  };
  auto tcvt = [&](const float* W, unsigned short* Wt_, int K, int N) {
    transpose_cvt<<<cdiv_ll((long long)K * N, 256), 256, 0, stream>>>(W, Wt_, K, N);
  };
  auto gemm = [&](const unsigned short* A, const unsigned short* Bt, float* C,
                  int M, int N, int K) {
    int tiles = (M >> 4) * (N >> 4);
    gemm_launch:
    wmma_gemm_bf16<<<cdiv_ll(tiles, 8), 256, 0, stream>>>(A, Bt, C, M, N, K);
    (void)0; goto gemm_done; gemm_done:;
  };

  auto relation = [&](const unsigned short* xs, int Ns, const unsigned short* xd, int Ndst,
                      const float* Ws, const float* Wd, const float* al, const float* ar,
                      const float* bias, const int* sI, const int* dI, int E, float* out) {
    tcvt(Ws, Wt, KDIM, FDIM);
    gemm(xs, Wt, fsbuf, Ns, FDIM, KDIM);
    const float* fsp = fsbuf;
    const float* fdp;
    if (Wd == Ws && xd == xs) {
      fdp = fsbuf;                                   // homogeneous relation: fd == fs
    } else {
      tcvt(Wd, Wt, KDIM, FDIM);
      gemm(xd, Wt, fdbuf, Ndst, FDIM, KDIM);
      fdp = fdbuf;
    }
    attn_score<<<cdiv_ll((long long)Ns * NHEAD, 256), 256, 0, stream>>>(fsp, al, el, Ns);
    attn_score<<<cdiv_ll((long long)Ndst * NHEAD, 256), 256, 0, stream>>>(fdp, ar, er, Ndst);
    hipMemsetAsync(emax, 0, (size_t)Ndst * NHEAD * 4, stream);
    hipMemsetAsync(den,  0, (size_t)Ndst * NHEAD * 4, stream);
    hipMemsetAsync(out,  0, (size_t)Ndst * FDIM * 4, stream);
    int g = cdiv_ll((long long)E * NHEAD, 256);
    edge_max <<<g, 256, 0, stream>>>(sI, dI, el, er, emax, E);
    edge_exp <<<g, 256, 0, stream>>>(sI, dI, el, er, emax, den, ee, E);
    edge_aggr<<<g, 256, 0, stream>>>(sI, dI, ee, den, fsp, out, E);
    bias_elu<<<cdiv_ll((long long)Ndst * FDIM, 256), 256, 0, stream>>>(out, bias, Ndst);
  };

  auto semantic = [&](const float* oA, const float* oB, const float* W1s, const float* b1s,
                      const float* w2s, float* outp) {
    cvt(oA, bfA, (long long)Nc * FDIM);
    cvt(oB, bfB, (long long)Nc * FDIM);
    tcvt(W1s, Wt, FDIM, 128);
    gemm(bfA, Wt, sA, Nc, 128, KDIM);
    gemm(bfB, Wt, sB, Nc, 128, KDIM);
    sem_combine<<<cdiv_ll(Nc, 256), 256, 0, stream>>>(sA, sB, b1s, w2s, oA, oB, outp, Nc);
  };

  const size_t WSZ = (size_t)256 * 256;  // one weight slice
  const int HD = NHEAD * DHEAD;

  // ===== layer 0 =====
  cvt(feat_drug, bfXd, (long long)Nd * FDIM);
  cvt(feat_cell, bfXc, (long long)Nc * FDIM);
  relation(bfXd, Nd, bfXd, Nd, Wsrc + 0 * WSZ, Wsrc + 0 * WSZ,
           attn_l + 0 * HD, attn_r + 0 * HD, gat_b + 0 * FDIM, src_dd, dst_dd, Edd, o_dd);
  relation(bfXd, Nd, bfXc, Nc, Wsrc + 1 * WSZ, Wdst_dc + 0 * WSZ,
           attn_l + 1 * HD, attn_r + 1 * HD, gat_b + 1 * FDIM, src_dc, dst_dc, Edc, o_dc);
  relation(bfXc, Nc, bfXc, Nc, Wsrc + 2 * WSZ, Wsrc + 2 * WSZ,
           attn_l + 2 * HD, attn_r + 2 * HD, gat_b + 2 * FDIM, src_cc, dst_cc, Ecc, o_cc);
  // drug semantic attention has M=1 -> identity: hd1 = o_dd (exact)
  semantic(o_dc, o_cc, sem_W1 + 1 * (FDIM * 128), sem_b1 + 1 * 128, sem_W2 + 1 * 128, hc1);

  // ===== layer 1 ===== (dd relation & drug semantic attn don't reach the output -> skipped)
  cvt(o_dd, bfXd, (long long)Nd * FDIM);
  cvt(hc1,  bfXc, (long long)Nc * FDIM);
  relation(bfXd, Nd, bfXc, Nc, Wsrc + 4 * WSZ, Wdst_dc + 1 * WSZ,
           attn_l + 4 * HD, attn_r + 4 * HD, gat_b + 4 * FDIM, src_dc, dst_dc, Edc, o_dc);
  relation(bfXc, Nc, bfXc, Nc, Wsrc + 5 * WSZ, Wsrc + 5 * WSZ,
           attn_l + 5 * HD, attn_r + 5 * HD, gat_b + 5 * FDIM, src_cc, dst_cc, Ecc, o_cc);
  semantic(o_dc, o_cc, sem_W1 + 3 * (FDIM * 128), sem_b1 + 3 * 128, sem_W2 + 3 * 128, hc2);

  // ===== DNN head =====
  cvt(hc2, bfA, (long long)Nc * FDIM);
  tcvt(dnn_W1, Wt, KDIM, 32);
  gemm(bfA, Wt, g1, Nc, 32, KDIM);
  head_kernel<<<cdiv_ll(Nc, 256), 256, 0, stream>>>(g1, dnn_b1, dnn_W2, dnn_b2,
                                                    dnn_W3, dnn_b3, (float*)d_out, Nc);
}
